// MLPNodeLink_87514253624020
// MI455X (gfx1250) — matrix operand order repaired and checked by reading, not compiled, not run on hardware
//
#include <hip/hip_runtime.h>
#include <hip/hip_bf16.h>

// ---------------------------------------------------------------------------
// N1=N2=512, F=128, H=512. pairs = [V1_i | V2_j] (262144 x 256), 3-layer MLP.
// Key structure: rows r = i*512 + j; a 128-row tile has constant i, so
//   pairs @ W1 = V1[i] @ W1[0:128,:]  (rank-1 shared part, f32, once per WG)
//              + V2[j] @ W1[128:256,:] (bf16 WMMA, K=128)
// Layers 1-2 via v_wmma_f32_16x16x32_bf16 (f32 accum); layer 3 (512->2) f32.
// Intermediates stay in LDS: HBM sees only ~3 MB of inputs + 2 MB output.
// ---------------------------------------------------------------------------

#define F_   128
#define H_   512
#define N2_  512
#define MT_  128   // rows of the pair matrix per workgroup

typedef __bf16 v16bf __attribute__((ext_vector_type(16)));
typedef float  v8f   __attribute__((ext_vector_type(8)));

union FragU { v16bf f; uint4 q[2]; };   // 32-byte bf16 WMMA operand fragment

// ---------------------------------------------------------------------------
// Pass 1: bf16 conversion (native cvt via __bf16 cast).
//   v2bf[512*128]            : V2 rows
//   w1b [512*128]  (N x K)   : transpose of W1[128:256, :]  (bottom half)
//   w2t [512*512]  (N x K)   : transpose of W2
// Total = 768 KB of workspace.
// ---------------------------------------------------------------------------
__global__ __launch_bounds__(256)
void convert_kernel(const float* __restrict__ V2,
                    const float* __restrict__ W1, const float* __restrict__ W2,
                    __bf16* __restrict__ ws)
{
  const int idx = blockIdx.x * 256 + threadIdx.x;
  __bf16* v2bf = ws;
  __bf16* w1b  = ws + 65536;    // 512 x 128 (N x K), K indexes W1 rows 128..255
  __bf16* w2t  = ws + 131072;   // 512 x 512 (N x K)
  if (idx < 65536) {
    v2bf[idx] = (__bf16)V2[idx];
  } else if (idx < 131072) {
    const int e = idx - 65536;                 // e = n*128 + k2
    const int n = e >> 7, k2 = e & 127;
    w1b[e] = (__bf16)W1[(F_ + k2) * H_ + n];   // W1 is (256 x 512) row-major
  } else if (idx < 393216) {
    const int e = idx - 131072;                // e = n*512 + k
    const int n = e >> 9, k = e & 511;
    w2t[e] = (__bf16)W2[k * H_ + n];           // W2 is (512 x 512) row-major
  }
}

// ---------------------------------------------------------------------------
// Pass 2: fused MLP. One block = 128 rows, 8 waves; each wave owns a
// 64-column N-slice. LDS: u (2KB f32) + H1 (128KB bf16) + H2 (128KB bf16).
// ---------------------------------------------------------------------------
__global__ __launch_bounds__(256)
void mlp_fused_kernel(const float* __restrict__ V1,
                      const __bf16* __restrict__ v2bf,
                      const __bf16* __restrict__ w1b,
                      const __bf16* __restrict__ w2t,
                      const float* __restrict__ W1,
                      const float* __restrict__ b1, const float* __restrict__ b2,
                      const float* __restrict__ w3, const float* __restrict__ b3,
                      float* __restrict__ out)
{
  extern __shared__ char smem[];
  float*  u  = reinterpret_cast<float*>(smem);                    // 512 f32
  __bf16* h1 = reinterpret_cast<__bf16*>(smem + 2048);            // 128x512
  __bf16* h2 = reinterpret_cast<__bf16*>(smem + 2048 + 262144);   // 128x512

  const int tid   = threadIdx.x;
  const int lane  = tid & 31;
  const int wave  = tid >> 5;
  const int lhalf = lane >> 4;            // 0 for lanes 0-15, 1 for 16-31
  const int lm    = lane & 15;
  const int r0    = blockIdx.x * MT_;
  const int i_idx = r0 >> 9;              // V1 row (constant across the tile)
  const int j0    = r0 & (N2_ - 1);       // first V2 row of the tile
  const int nbase = wave * 64;            // 8 waves x 64 cols = 512

  // -------- Phase 0: shared rank-1 part u = V1[i] @ W1[0:128,:] (f32) ------
  {
    const float* v1row = V1 + i_idx * F_;           // uniform -> scalar loads
    float s0 = 0.0f, s1 = 0.0f;
#pragma unroll 8
    for (int k = 0; k < F_; ++k) {
      const float a = v1row[k];
      s0 = fmaf(a, W1[k * H_ + tid], s0);
      s1 = fmaf(a, W1[k * H_ + tid + 256], s1);
    }
    u[tid]       = s0;
    u[tid + 256] = s1;
  }
  __syncthreads();

  // -------- Layer 1: H1 = relu(V2[j] @ W1_bot + (u + b1)), K = 128 ---------
  for (int mt = 0; mt < 8; ++mt) {
    const int m0 = mt * 16;
    v8f acc[4];
#pragma unroll
    for (int nt = 0; nt < 4; ++nt) acc[nt] = (v8f)(0.0f);

    const __bf16* rowA = v2bf + (j0 + m0 + lm) * F_;   // A row for this lane

#pragma unroll
    for (int k0 = 0; k0 < F_; k0 += 32) {
      // A fragment (16x32 bf16): per-lane two 16B chunks at k0+{0|8} and +16
      FragU a;
      const int k1 = k0 + lhalf * 8;
      a.q[0] = *reinterpret_cast<const uint4*>(rowA + k1);
      a.q[1] = *reinterpret_cast<const uint4*>(rowA + k1 + 16);

      const int kk = k0 + lhalf * 8;      // B fragment K base for this lane
#pragma unroll
      for (int nt = 0; nt < 4; ++nt) {
        FragU b;                          // B (32x16): lane = col, K chunks at kk and kk+16
        const __bf16* pb = w1b + (nbase + nt * 16 + lm) * F_ + kk;
        b.q[0] = *reinterpret_cast<const uint4*>(pb);
        b.q[1] = *reinterpret_cast<const uint4*>(pb + 16);
        acc[nt] = __builtin_amdgcn_wmma_f32_16x16x32_bf16(
            false, a.f, false, b.f, (short)0, acc[nt], false, false);
      }
    }

    // epilogue: (u + b1) + ReLU + bf16 -> LDS H1
    const int rbase = m0 + lhalf * 8;     // C/D: VGPR v -> row m0+v+8*lhalf
#pragma unroll
    for (int nt = 0; nt < 4; ++nt) {
      const int n = nbase + nt * 16 + lm;
      const float bias = b1[n] + u[n];
#pragma unroll
      for (int v = 0; v < 8; ++v) {
        h1[(rbase + v) * H_ + n] = (__bf16)fmaxf(acc[nt][v] + bias, 0.0f);
      }
    }
  }
  __syncthreads();

  // -------- Layer 2: H2 = relu(H1 @ W2 + b2), K = 512 ----------------------
  for (int mt = 0; mt < 8; ++mt) {
    const int m0 = mt * 16;
    v8f acc[4];
#pragma unroll
    for (int nt = 0; nt < 4; ++nt) acc[nt] = (v8f)(0.0f);

    const __bf16* arow = h1 + (m0 + lm) * H_;

#pragma unroll 4
    for (int k0 = 0; k0 < H_; k0 += 32) {
      FragU a;
      const int k1 = k0 + lhalf * 8;
      a.q[0] = *reinterpret_cast<const uint4*>(arow + k1);        // ds_load_b128
      a.q[1] = *reinterpret_cast<const uint4*>(arow + k1 + 16);
      const int kk = k0 + lhalf * 8;
#pragma unroll
      for (int nt = 0; nt < 4; ++nt) {
        FragU b;
        const __bf16* pb = w2t + (nbase + nt * 16 + lm) * H_ + kk;
        b.q[0] = *reinterpret_cast<const uint4*>(pb);
        b.q[1] = *reinterpret_cast<const uint4*>(pb + 16);
        acc[nt] = __builtin_amdgcn_wmma_f32_16x16x32_bf16(
            false, a.f, false, b.f, (short)0, acc[nt], false, false);
      }
    }

    const int rbase = m0 + lhalf * 8;
#pragma unroll
    for (int nt = 0; nt < 4; ++nt) {
      const int n = nbase + nt * 16 + lm;
      const float bias = b2[n];
#pragma unroll
      for (int v = 0; v < 8; ++v) {
        h2[(rbase + v) * H_ + n] = (__bf16)fmaxf(acc[nt][v] + bias, 0.0f);
      }
    }
  }
  __syncthreads();

  // -------- Layer 3: out = relu(H2 @ W3 + b3), W3 is 512x2 -----------------
  // 256 threads: thread t handles (row = t>>1, output col = t&1).
  {
    const int rrow = tid >> 1;
    const int c    = tid & 1;
    const __bf16* hr = h2 + rrow * H_;
    float s = 0.0f;
#pragma unroll 4
    for (int k = 0; k < H_; k += 8) {
      const uint4 q = *reinterpret_cast<const uint4*>(hr + k);
      const unsigned uu[4] = {q.x, q.y, q.z, q.w};
#pragma unroll
      for (int e = 0; e < 4; ++e) {
        const float lo = __uint_as_float(uu[e] << 16);
        const float hi = __uint_as_float(uu[e] & 0xFFFF0000u);
        s = fmaf(lo, w3[(k + 2 * e    ) * 2 + c], s);
        s = fmaf(hi, w3[(k + 2 * e + 1) * 2 + c], s);
      }
    }
    out[(r0 + rrow) * 2 + c] = fmaxf(s + b3[c], 0.0f);
  }
}

// ---------------------------------------------------------------------------
extern "C" void kernel_launch(void* const* d_in, const int* in_sizes, int n_in,
                              void* d_out, int out_size, void* d_ws, size_t ws_size,
                              hipStream_t stream) {
  (void)in_sizes; (void)n_in; (void)out_size; (void)ws_size;
  const float* V1 = (const float*)d_in[0];   // (1,512,128)
  const float* V2 = (const float*)d_in[1];   // (1,512,128)
  const float* W1 = (const float*)d_in[2];   // (256,512)
  const float* b1 = (const float*)d_in[3];   // (512)
  const float* W2 = (const float*)d_in[4];   // (512,512)
  const float* b2 = (const float*)d_in[5];   // (512)
  const float* W3 = (const float*)d_in[6];   // (512,2)
  const float* b3 = (const float*)d_in[7];   // (2)
  float* out = (float*)d_out;
  __bf16* ws = (__bf16*)d_ws;                // needs 768 KB

  // Pass 1: bf16 conversion + weight transposes (393216 elements).
  convert_kernel<<<393216 / 256, 256, 0, stream>>>(V2, W1, W2, ws);

  const __bf16* v2bf = ws;
  const __bf16* w1b  = ws + 65536;
  const __bf16* w2t  = ws + 131072;

  // Pass 2: fused MLP. 262144 rows / 128 per block = 2048 blocks.
  const size_t lds_bytes = 2048 + 2 * (size_t)MT_ * H_ * sizeof(__bf16); // 258 KB
  mlp_fused_kernel<<<(512 * 512) / MT_, 256, lds_bytes, stream>>>(
      V1, v2bf, w1b, w2t, W1, b1, b2, W3, b3, out);
}
